// lower_level_575525618299
// MI455X (gfx1250) — matrix operand order repaired and checked by reading, not compiled, not run on hardware
//
#include <hip/hip_runtime.h>

typedef __attribute__((ext_vector_type(16))) __bf16 v16bf;
typedef __attribute__((ext_vector_type(8)))  float  v8f;

#define NDIM 8192
#define KDIM 64
#define JSPLIT 4
#define JB_TOTAL (NDIM / 32)          /* 256 j-blocks of 32 */
#define JB_PER   (JB_TOTAL / JSPLIT)  /* 64 per block */
#define WAVES_PER_BLOCK 8
#define ROWGROUPS (NDIM / 16 / WAVES_PER_BLOCK) /* 64 */
#define BPACK_ELEMS ((size_t)JB_TOTAL * 4 * 32 * 16) /* 524288 bf16 = 1 MB */

// ---------------------------------------------------------------- zero init
__global__ void k_zero(float* __restrict__ out, float* __restrict__ G) {
    int g = blockIdx.x * blockDim.x + threadIdx.x;
    if (g < KDIM * KDIM) G[g] = 0.0f;
    if (g == 0) out[0] = 0.0f;
}

// ----------------------------------------------- pre-pack F_ll as bf16 B-op
// B operand packing used here (16x16x32 bf16, wave32): lane l holds column N = l%16,
// K-values khalf*16 .. khalf*16+15 contiguously (khalf = l/16), 2 per VGPR.
__global__ void k_pack_b(const float* __restrict__ Fll, __bf16* __restrict__ Bp) {
    int g = blockIdx.x * blockDim.x + threadIdx.x;   // 0 .. 32767
    int lane  = g & 31;
    int kt    = (g >> 5) & 3;
    int jb    = g >> 7;
    int n     = lane & 15;
    int khalf = lane >> 4;
    int col   = kt * 16 + n;
    int j0    = jb * 32 + khalf * 16;
    __bf16* dst = Bp + (size_t)g * 16;
#pragma unroll
    for (int k = 0; k < 16; ++k)
        dst[k] = (__bf16)Fll[(size_t)(j0 + k) * KDIM + col];
}

// ------------------------------------------ term1 = sum((Theta@F_ll)*F_ll)
__global__ void __launch_bounds__(256)
k_term1(const float* __restrict__ Theta, const float* __restrict__ Fll,
        const __bf16* __restrict__ Bp, float* __restrict__ out) {
    __shared__ float red[256];
    int tid  = threadIdx.x;
    int lane = tid & 31;
    int wave = tid >> 5;
    int jchunk   = blockIdx.x % JSPLIT;
    int rowgroup = blockIdx.x / JSPLIT;
    int iBase = (rowgroup * WAVES_PER_BLOCK + wave) * 16;
    int m     = lane & 15;
    int khalf = lane >> 4;
    const float* trow = Theta + (size_t)(iBase + m) * NDIM;

    v8f c0 = {}, c1 = {}, c2 = {}, c3 = {};

    int jb0 = jchunk * JB_PER;
    for (int jb = jb0; jb < jb0 + JB_PER; ++jb) {
        int j0 = jb * 32;
        // A operand: 16 f32 -> bf16, contiguous K order matching packed B:
        // lanes 0-15: K=0..15; lanes 16-31: K=16..31
        const float4* pa = (const float4*)(trow + j0 + khalf * 16);
        float4 f0 = pa[0], f1 = pa[1];
        const float4* pb = (const float4*)(trow + j0 + khalf * 16 + 8);
        float4 g0 = pb[0], g1 = pb[1];
        __builtin_prefetch(trow + j0 + 1024, 0, 3);  // stream Theta ~4 KB ahead

        v16bf a;
        a[0]  = (__bf16)f0.x; a[1]  = (__bf16)f0.y; a[2]  = (__bf16)f0.z; a[3]  = (__bf16)f0.w;
        a[4]  = (__bf16)f1.x; a[5]  = (__bf16)f1.y; a[6]  = (__bf16)f1.z; a[7]  = (__bf16)f1.w;
        a[8]  = (__bf16)g0.x; a[9]  = (__bf16)g0.y; a[10] = (__bf16)g0.z; a[11] = (__bf16)g0.w;
        a[12] = (__bf16)g1.x; a[13] = (__bf16)g1.y; a[14] = (__bf16)g1.z; a[15] = (__bf16)g1.w;

        const v16bf* bbase = (const v16bf*)(Bp + ((size_t)jb * 4 * 32) * 16);
        v16bf b0 = bbase[0 * 32 + lane];
        v16bf b1 = bbase[1 * 32 + lane];
        v16bf b2 = bbase[2 * 32 + lane];
        v16bf b3 = bbase[3 * 32 + lane];

        c0 = __builtin_amdgcn_wmma_f32_16x16x32_bf16(false, a, false, b0, (short)0, c0, false, false);
        c1 = __builtin_amdgcn_wmma_f32_16x16x32_bf16(false, a, false, b1, (short)0, c1, false, false);
        c2 = __builtin_amdgcn_wmma_f32_16x16x32_bf16(false, a, false, b2, (short)0, c2, false, false);
        c3 = __builtin_amdgcn_wmma_f32_16x16x32_bf16(false, a, false, b3, (short)0, c3, false, false);
    }

    // C/D layout: VGPR r -> row M = r + 8*(lane/16), col N = lane%16
    float s = 0.0f;
    int n  = lane & 15;
    int hi = lane >> 4;
#pragma unroll
    for (int r = 0; r < 8; ++r) {
        size_t row = (size_t)(iBase + r + 8 * hi) * KDIM;
        s += c0[r] * Fll[row +  0 + n];
        s += c1[r] * Fll[row + 16 + n];
        s += c2[r] * Fll[row + 32 + n];
        s += c3[r] * Fll[row + 48 + n];
    }
    red[tid] = s;
    __syncthreads();
    for (int off = 128; off > 0; off >>= 1) {
        if (tid < off) red[tid] += red[tid + off];
        __syncthreads();
    }
    if (tid == 0) atomicAdd(out, red[0]);
}

// ----------------------------------- G[ky][kx] += sum_r F_ul[r,ky]*F_ll[r,kx]
__global__ void k_gram(const float* __restrict__ Ful, const float* __restrict__ Fll,
                       float* __restrict__ G) {
    int tid = threadIdx.x;
    int ky  = tid >> 2;           // 0..63
    int kx0 = (tid & 3) * 16;     // 0,16,32,48
    int r0  = blockIdx.x * 128;
    float acc[16];
#pragma unroll
    for (int i = 0; i < 16; ++i) acc[i] = 0.0f;
    for (int r = r0; r < r0 + 128; ++r) {
        float u = Ful[(size_t)r * KDIM + ky];
        const float4* p = (const float4*)(Fll + (size_t)r * KDIM + kx0);
        float4 x0 = p[0], x1 = p[1], x2 = p[2], x3 = p[3];
        acc[0]  += u * x0.x; acc[1]  += u * x0.y; acc[2]  += u * x0.z; acc[3]  += u * x0.w;
        acc[4]  += u * x1.x; acc[5]  += u * x1.y; acc[6]  += u * x1.z; acc[7]  += u * x1.w;
        acc[8]  += u * x2.x; acc[9]  += u * x2.y; acc[10] += u * x2.z; acc[11] += u * x2.w;
        acc[12] += u * x3.x; acc[13] += u * x3.y; acc[14] += u * x3.z; acc[15] += u * x3.w;
    }
#pragma unroll
    for (int i = 0; i < 16; ++i) atomicAdd(&G[ky * KDIM + kx0 + i], acc[i]);
}

// ------------------------------------------------- out += lambda * sum(G*G)
__global__ void k_term2(const float* __restrict__ G, const float* __restrict__ lam,
                        float* __restrict__ out) {
    __shared__ float red[256];
    int tid = threadIdx.x;
    float s = 0.0f;
    for (int i = tid; i < KDIM * KDIM; i += 256) { float g = G[i]; s += g * g; }
    red[tid] = s;
    __syncthreads();
    for (int off = 128; off > 0; off >>= 1) {
        if (tid < off) red[tid] += red[tid + off];
        __syncthreads();
    }
    if (tid == 0) atomicAdd(out, lam[0] * red[0]);
}

extern "C" void kernel_launch(void* const* d_in, const int* in_sizes, int n_in,
                              void* d_out, int out_size, void* d_ws, size_t ws_size,
                              hipStream_t stream) {
    const float* Fll   = (const float*)d_in[0];
    const float* Ful   = (const float*)d_in[1];
    const float* Theta = (const float*)d_in[2];
    const float* lam   = (const float*)d_in[3];
    float* out = (float*)d_out;

    __bf16* Bp = (__bf16*)d_ws;
    float*  G  = (float*)((char*)d_ws + BPACK_ELEMS * sizeof(__bf16)); // +1 MB

    k_zero<<<16, 256, 0, stream>>>(out, G);
    k_pack_b<<<(JB_TOTAL * 4 * 32) / 256, 256, 0, stream>>>(Fll, Bp);
    k_term1<<<ROWGROUPS * JSPLIT, 256, 0, stream>>>(Theta, Fll, Bp, out);
    k_gram<<<NDIM / 128, 256, 0, stream>>>(Ful, Fll, G);
    k_term2<<<1, 256, 0, stream>>>(G, lam, out);
}
